// RandomizedAttentionPolicy_84567906058447
// MI455X (gfx1250) — matrix operand order repaired and checked by reading, not compiled, not run on hardware
//
#include <hip/hip_runtime.h>
#include <hip/hip_bf16.h>
#include <math.h>

// ---------------------------------------------------------------------------
// RandomizedAttentionPolicy forward for MI455X (gfx1250, wave32, WMMA).
// All large matmuls use v_wmma_f32_16x16x32_f16 (f16 A/B, f32 accum).
// ---------------------------------------------------------------------------

typedef __attribute__((ext_vector_type(16))) _Float16 v16h;
typedef __attribute__((ext_vector_type(8)))  _Float16 v8h;
typedef __attribute__((ext_vector_type(8)))  float    v8f;

#define Bv   1024
#define Lv   64
#define Hv   256
#define ACTv 8
#define INv  24
#define INP  32          // padded K for first layer
#define ML   (Bv * Lv)   // 65536 rows of the embedding GEMMs

// ---- WMMA fragment helpers -------------------------------------------------
// A fragment: 16x32 (MxK) f16, A row-major with stride lda.
//   lanes 0-15 : row=lane,    K = {0..7, 16..23}
//   lanes 16-31: row=lane-16, K = {8..15, 24..31}
__device__ __forceinline__ v16h load_afrag(const _Float16* __restrict__ A, int lda) {
  int lane = threadIdx.x & 31;
  int row  = lane & 15;
  int kb   = (lane & 16) >> 1;       // 0 or 8
  const _Float16* p = A + (size_t)row * lda + kb;
  v8h lo = *(const v8h*)(p);
  v8h hi = *(const v8h*)(p + 16);
  v16h f;
#pragma unroll
  for (int i = 0; i < 8; ++i) { f[i] = lo[i]; f[8 + i] = hi[i]; }
  return f;
}

// B fragment: 32x16 (KxN) f16, B row-major [K][N] with stride ldb.
//   lane = N-col (lane & 15), K = {0..7, 16..23} / {8..15, 24..31} per lane half.
__device__ __forceinline__ v16h load_bfrag(const _Float16* __restrict__ B, int ldb) {
  int lane = threadIdx.x & 31;
  int n  = lane & 15;
  int kb = (lane & 16) >> 1;
  v16h f;
#pragma unroll
  for (int i = 0; i < 8; ++i) {
    f[i]     = B[(size_t)(kb + i) * ldb + n];
    f[8 + i] = B[(size_t)(16 + kb + i) * ldb + n];
  }
  return f;
}

__device__ __forceinline__ v8f wmma16(v16h a, v16h b, v8f c) {
  return __builtin_amdgcn_wmma_f32_16x16x32_f16(false, a, false, b, (short)0, c,
                                                false, false);
}

// ---- data prep kernels -----------------------------------------------------
__global__ void obs_pad_k(const float* __restrict__ obs, _Float16* __restrict__ obsh) {
  int i = blockIdx.x * blockDim.x + threadIdx.x;   // ML * INP
  if (i >= ML * INP) return;
  int r = i >> 5, c = i & 31;
  obsh[i] = (c < INv) ? (_Float16)obs[r * INv + c] : (_Float16)0.f;
}

// w [N=fout][K=fin] f32  ->  wT [Kp][N] f16 (zero-pad K to Kp)
__global__ void wconv_t_k(const float* __restrict__ w, _Float16* __restrict__ wt,
                          int N, int K, int Kp) {
  int i = blockIdx.x * blockDim.x + threadIdx.x;
  if (i >= Kp * N) return;
  int k = i / N, n = i % N;
  wt[i] = (k < K) ? (_Float16)w[n * K + k] : (_Float16)0.f;
}

// ---- generic WMMA GEMM: C[M,N] = act(A[M,K] @ B[K,N] + bias) ---------------
template <bool RELU, bool STORE_H, bool STORE_F>
__global__ void gemm_k(const _Float16* __restrict__ A, const _Float16* __restrict__ B,
                       const float* __restrict__ bias,
                       _Float16* __restrict__ Ch, float* __restrict__ Cf,
                       int M, int N, int K) {
  int wave = (blockIdx.x * blockDim.x + threadIdx.x) >> 5;
  int ntn = N >> 4;
  int total = (M >> 4) * ntn;
  if (wave >= total) return;                       // wave-uniform guard
  int mt = wave / ntn, nt = wave % ntn;
  int n0 = nt << 4;
  const _Float16* Ab = A + (size_t)(mt << 4) * K;
  v8f acc = {};
  for (int k = 0; k < K; k += 32) {
    v16h af = load_afrag(Ab + k, K);
    v16h bf = load_bfrag(B + (size_t)k * N + n0, N);
    acc = wmma16(af, bf, acc);
  }
  int lane = threadIdx.x & 31;
  int col = lane & 15;
  int rb  = (lane & 16) >> 1;
  float bv = bias[n0 + col];
#pragma unroll
  for (int r = 0; r < 8; ++r) {
    float v = acc[r] + bv;
    if (RELU) v = fmaxf(v, 0.f);
    size_t off = (size_t)((mt << 4) + rb + r) * N + n0 + col;
    if constexpr (STORE_H) Ch[off] = (_Float16)v;
    if constexpr (STORE_F) Cf[off] = v;
  }
}

// ---- fused score GEMM + log-softmax(L) + landmark mean ---------------------
// One block = one (batch b, 16-col chunk): 4 waves compute the 64x16 score
// tile (emb @ isw^T + isb) via WMMA into LDS, then log-softmax over L=64 and
// mean_L(emb + score) -> message[b, 16 cols] (== temp_state.mean).
__global__ void score_message_k(const _Float16* __restrict__ embh,
                                const _Float16* __restrict__ iswT,
                                const float* __restrict__ isb,
                                float* __restrict__ message,
                                _Float16* __restrict__ messageh) {
  __shared__ float stile[Lv][16];
  int b  = blockIdx.x >> 4;
  int nc = blockIdx.x & 15;
  int n0 = nc << 4;
  int wv = threadIdx.x >> 5;                       // 0..3 -> 16-row slab
  const _Float16* Ab = embh + (size_t)(b * Lv + wv * 16) * Hv;
  v8f acc = {};
  for (int k = 0; k < Hv; k += 32) {
    v16h af = load_afrag(Ab + k, Hv);
    v16h bf = load_bfrag(iswT + (size_t)k * Hv + n0, Hv);
    acc = wmma16(af, bf, acc);
  }
  int lane = threadIdx.x & 31;
  int col = lane & 15;
  int rb  = (lane & 16) >> 1;
  float bv = isb[n0 + col];
#pragma unroll
  for (int r = 0; r < 8; ++r)
    stile[wv * 16 + rb + r][col] = acc[r] + bv;
  __syncthreads();
  if (threadIdx.x < 16) {
    int c = threadIdx.x;
    float mx = -1e30f;
    for (int l = 0; l < Lv; ++l) mx = fmaxf(mx, stile[l][c]);
    float se = 0.f;
    for (int l = 0; l < Lv; ++l) se += __expf(stile[l][c] - mx);
    float lse = mx + __logf(se);
    float s = 0.f;
    for (int l = 0; l < Lv; ++l) {
      float e = (float)embh[(size_t)(b * Lv + l) * Hv + n0 + c];
      s += e + stile[l][c] - lse;
    }
    float m = s * (1.f / (float)Lv);
    message[b * Hv + n0 + c]  = m;
    messageh[b * Hv + n0 + c] = (_Float16)m;
  }
}

// ---- consensus mask machinery ---------------------------------------------
__global__ void mask_k(const float* __restrict__ rnd, const int* __restrict__ idx,
                       _Float16* __restrict__ maskh) {
  int i = blockIdx.x * blockDim.x + threadIdx.x;   // Bv*Bv
  if (i >= Bv * Bv) return;
  int r = i >> 10, c = i & 1023;
  maskh[i] = (rnd[idx[c]] >= rnd[idx[r]]) ? (_Float16)1.f : (_Float16)0.f;
}

__global__ void cnt_k(const float* __restrict__ rnd, const int* __restrict__ idx,
                      float* __restrict__ cnt) {
  int i = blockIdx.x * blockDim.x + threadIdx.x;
  if (i >= Bv) return;
  float ri = rnd[idx[i]];
  int c = 0;
  for (int j = 0; j < Bv; ++j) c += (rnd[idx[j]] >= ri) ? 1 : 0;
  cnt[i] = (float)c;
}

__global__ void colmax_k(const float* __restrict__ scores, float* __restrict__ cm) {
  int h = threadIdx.x;                             // 256 threads, 1 block
  float m = -1e30f;
  for (int j = 0; j < Bv; ++j) m = fmaxf(m, scores[(size_t)j * Hv + h]);
  cm[h] = m;
}

__global__ void exp_pack_k(const float* __restrict__ scores,
                           const float* __restrict__ msgs,
                           const float* __restrict__ cm,
                           _Float16* __restrict__ E, _Float16* __restrict__ P) {
  int i = blockIdx.x * blockDim.x + threadIdx.x;   // Bv*Hv, [j][h] row-major
  if (i >= Bv * Hv) return;
  int h = i & 255;
  float sc = scores[i];
  E[i] = (_Float16)__expf(sc - cm[h]);
  P[i] = (_Float16)(msgs[i] + sc);
}

// S = mask @ E, T = mask @ P  (shared A fragment, two accumulators)
__global__ void mask_gemm2_k(const _Float16* __restrict__ maskh,
                             const _Float16* __restrict__ E,
                             const _Float16* __restrict__ P,
                             float* __restrict__ S, float* __restrict__ T) {
  int wave = (blockIdx.x * blockDim.x + threadIdx.x) >> 5;
  if (wave >= (Bv >> 4) * (Hv >> 4)) return;
  int mt = wave >> 4, nt = wave & 15;
  int n0 = nt << 4;
  const _Float16* Ab = maskh + (size_t)(mt << 4) * Bv;
  v8f s = {}, t = {};
  for (int k = 0; k < Bv; k += 32) {
    v16h af = load_afrag(Ab + k, Bv);
    v16h b1 = load_bfrag(E + (size_t)k * Hv + n0, Hv);
    v16h b2 = load_bfrag(P + (size_t)k * Hv + n0, Hv);
    s = wmma16(af, b1, s);
    t = wmma16(af, b2, t);
  }
  int lane = threadIdx.x & 31;
  int col = lane & 15;
  int rb  = (lane & 16) >> 1;
#pragma unroll
  for (int r = 0; r < 8; ++r) {
    size_t off = (size_t)((mt << 4) + rb + r) * Hv + n0 + col;
    S[off] = s[r];
    T[off] = t[r];
  }
}

__global__ void combine_k(const float* __restrict__ message,
                          const float* __restrict__ T, const float* __restrict__ S,
                          const float* __restrict__ cnt, const float* __restrict__ cm,
                          float* __restrict__ outb) {
  int i = blockIdx.x * blockDim.x + threadIdx.x;
  if (i >= Bv * Hv) return;
  int row = i >> 8, h = i & 255;
  float lse = __logf(S[i]) + cm[h];
  outb[i] = message[i] + T[i] - cnt[row] * lse;
}

// ---- heads + tanh-normal sample + log-prob ---------------------------------
__global__ void final_k(const float* __restrict__ outb,
                        const float* __restrict__ mw, const float* __restrict__ mb,
                        const float* __restrict__ sw, const float* __restrict__ sb,
                        const float* __restrict__ eps, float* __restrict__ out) {
  int b = blockIdx.x * blockDim.x + threadIdx.x;
  if (b >= Bv) return;
  const float* x = outb + (size_t)b * Hv;
  float lp = 0.f;
#pragma unroll 1
  for (int a = 0; a < ACTv; ++a) {
    float m  = mb[a];
    float ls = sb[a];
    for (int h = 0; h < Hv; ++h) {
      float xv = x[h];
      m  += xv * mw[a * Hv + h];
      ls += xv * sw[a * Hv + h];
    }
    float sd  = __expf(ls);
    float pre = m + sd * eps[b * ACTv + a];
    out[b * ACTv + a] = tanhf(pre);
    float var = sd * sd;
    float d   = pre - m;
    float lpg = -0.5f * (d * d / (var + 1e-8f) + 2.f * ls + 1.8378770664093453f);
    float xn  = -2.f * pre;
    float sp  = (xn > 20.f) ? xn : log1pf(__expf(xn));
    float cor = 2.f * (0.6931471805599453f - pre - sp);
    lp += lpg - cor;
  }
  out[Bv * ACTv + b] = lp;
}

// ---------------------------------------------------------------------------
extern "C" void kernel_launch(void* const* d_in, const int* in_sizes, int n_in,
                              void* d_out, int out_size, void* d_ws, size_t ws_size,
                              hipStream_t stream) {
  const float* obs = (const float*)d_in[0];
  const float* rnd = (const float*)d_in[1];
  const float* eps = (const float*)d_in[2];
  const float* e1w = (const float*)d_in[3];  const float* e1b = (const float*)d_in[4];
  const float* e2w = (const float*)d_in[5];  const float* e2b = (const float*)d_in[6];
  const float* isw = (const float*)d_in[7];  const float* isb = (const float*)d_in[8];
  const float* c1w = (const float*)d_in[9];  const float* c1b = (const float*)d_in[10];
  const float* c2w = (const float*)d_in[11]; const float* c2b = (const float*)d_in[12];
  const float* csw = (const float*)d_in[13]; const float* csb = (const float*)d_in[14];
  const float* mw  = (const float*)d_in[15]; const float* mb  = (const float*)d_in[16];
  const float* sw  = (const float*)d_in[17]; const float* sb  = (const float*)d_in[18];
  const int*   idx = (const int*)d_in[19];
  float* out = (float*)d_out;

  // workspace carve-up (256B aligned)
  char* ws = (char*)d_ws; size_t off = 0;
  auto carve = [&](size_t bytes) -> char* {
    char* p = ws + off; off += (bytes + 255) & ~(size_t)255; return p;
  };
  _Float16* obsh  = (_Float16*)carve((size_t)ML * INP * 2);
  _Float16* e1wT  = (_Float16*)carve((size_t)INP * Hv * 2);
  _Float16* e2wT  = (_Float16*)carve((size_t)Hv * Hv * 2);
  _Float16* iswT  = (_Float16*)carve((size_t)Hv * Hv * 2);
  _Float16* c1wT  = (_Float16*)carve((size_t)Hv * Hv * 2);
  _Float16* c2wT  = (_Float16*)carve((size_t)Hv * Hv * 2);
  _Float16* cswT  = (_Float16*)carve((size_t)Hv * Hv * 2);
  _Float16* x1h   = (_Float16*)carve((size_t)ML * Hv * 2);
  _Float16* embh  = (_Float16*)carve((size_t)ML * Hv * 2);
  float*    msg   = (float*)   carve((size_t)Bv * Hv * 4);
  _Float16* msgh  = (_Float16*)carve((size_t)Bv * Hv * 2);
  _Float16* th    = (_Float16*)carve((size_t)Bv * Hv * 2);
  float*    msgs  = (float*)   carve((size_t)Bv * Hv * 4);
  _Float16* msgsh = (_Float16*)carve((size_t)Bv * Hv * 2);
  float*    scrs  = (float*)   carve((size_t)Bv * Hv * 4);
  float*    cmax  = (float*)   carve((size_t)Hv * 4);
  _Float16* maskh = (_Float16*)carve((size_t)Bv * Bv * 2);
  float*    cnt   = (float*)   carve((size_t)Bv * 4);
  _Float16* Eh    = (_Float16*)carve((size_t)Bv * Hv * 2);
  _Float16* Ph    = (_Float16*)carve((size_t)Bv * Hv * 2);
  float*    Sf    = (float*)   carve((size_t)Bv * Hv * 4);
  float*    Tf    = (float*)   carve((size_t)Bv * Hv * 4);
  float*    outb  = (float*)   carve((size_t)Bv * Hv * 4);

  const int BD = 256;
  // prep: pad obs, transpose+convert weights to f16 [K][N]
  obs_pad_k<<<(ML * INP + BD - 1) / BD, BD, 0, stream>>>(obs, obsh);
  wconv_t_k<<<(INP * Hv + BD - 1) / BD, BD, 0, stream>>>(e1w, e1wT, Hv, INv, INP);
  wconv_t_k<<<(Hv * Hv + BD - 1) / BD, BD, 0, stream>>>(e2w, e2wT, Hv, Hv, Hv);
  wconv_t_k<<<(Hv * Hv + BD - 1) / BD, BD, 0, stream>>>(isw, iswT, Hv, Hv, Hv);
  wconv_t_k<<<(Hv * Hv + BD - 1) / BD, BD, 0, stream>>>(c1w, c1wT, Hv, Hv, Hv);
  wconv_t_k<<<(Hv * Hv + BD - 1) / BD, BD, 0, stream>>>(c2w, c2wT, Hv, Hv, Hv);
  wconv_t_k<<<(Hv * Hv + BD - 1) / BD, BD, 0, stream>>>(csw, cswT, Hv, Hv, Hv);

  // embedding MLP (WMMA): x1 = relu(obs@e1w^T+b1); emb = relu(x1@e2w^T+b2)
  {
    int waves = (ML >> 4) * (Hv >> 4);
    int grid = (waves * 32 + BD - 1) / BD;
    gemm_k<true, true, false><<<grid, BD, 0, stream>>>(obsh, e1wT, e1b, x1h, nullptr,
                                                       ML, Hv, INP);
    gemm_k<true, true, false><<<grid, BD, 0, stream>>>(x1h, e2wT, e2b, embh, nullptr,
                                                       ML, Hv, Hv);
  }
  // fused score GEMM + log-softmax over L + landmark mean -> message
  score_message_k<<<Bv * (Hv / 16), 128, 0, stream>>>(embh, iswT, isb, msg, msgh);

  // consensus MLP (WMMA)
  {
    int waves = (Bv >> 4) * (Hv >> 4);
    int grid = (waves * 32 + BD - 1) / BD;
    gemm_k<true, true, false><<<grid, BD, 0, stream>>>(msgh, c1wT, c1b, th, nullptr,
                                                       Bv, Hv, Hv);
    gemm_k<false, true, true><<<grid, BD, 0, stream>>>(th, c2wT, c2b, msgsh, msgs,
                                                       Bv, Hv, Hv);
    gemm_k<false, false, true><<<grid, BD, 0, stream>>>(msgsh, cswT, csb, nullptr, scrs,
                                                        Bv, Hv, Hv);
  }

  // masked logsumexp via WMMA: S = mask@exp(scores-M), T = mask@(msgs+scores)
  mask_k<<<(Bv * Bv + BD - 1) / BD, BD, 0, stream>>>(rnd, idx, maskh);
  cnt_k<<<(Bv + BD - 1) / BD, BD, 0, stream>>>(rnd, idx, cnt);
  colmax_k<<<1, Hv, 0, stream>>>(scrs, cmax);
  exp_pack_k<<<(Bv * Hv + BD - 1) / BD, BD, 0, stream>>>(scrs, msgs, cmax, Eh, Ph);
  mask_gemm2_k<<<((Bv >> 4) * (Hv >> 4) * 32 + BD - 1) / BD, BD, 0, stream>>>(
      maskh, Eh, Ph, Sf, Tf);
  combine_k<<<(Bv * Hv + BD - 1) / BD, BD, 0, stream>>>(msg, Tf, Sf, cnt, cmax, outb);

  // heads + tanh-normal sample + log-prob
  final_k<<<(Bv + BD - 1) / BD, BD, 0, stream>>>(outb, mw, mb, sw, sb, eps, out);
}
